// WaffleIron_39737037423042
// MI455X (gfx1250) — matrix operand order repaired and checked by reading, not compiled, not run on hardware
//
#include <hip/hip_runtime.h>
#include <hip/hip_bf16.h>

// ---------------------------------------------------------------------------
// MI455X (gfx1250) implementation of the WaffleIron reference.
// Pipeline per layer d:
//   bn_stats -> fold BN into scale/shift -> zero grid -> scatter(atomic f32)
//   -> fused dwconv3x3+relu+dwconv3x3 (LDS tiled) -> gather+residual
//   -> bn_stats -> fold BN into W1/b1 (bf16 weights) -> WMMA GEMM1 (relu,bf16 h)
//   -> WMMA GEMM2 (f32 residual into tokens)
// GEMMs use v_wmma_f32_16x16x32_bf16 (wave32, 8 waves/block, 128x64 block tile)
// with the A (weight) tile staged into LDS by the Tensor Data Mover
// (tensor_load_to_lds + s_wait_tensorcnt), using D# hardware padding to
// produce the bank-conflict-avoiding 40-bf16 LDS row pitch directly.
// ---------------------------------------------------------------------------

typedef __attribute__((ext_vector_type(16))) __bf16 v16bf;
typedef __attribute__((ext_vector_type(8)))  float  v8f;

static constexpr int BB  = 2;
static constexpr int CC  = 256;     // channels (also GEMM M and K)
static constexpr int NN  = 32768;   // tokens per batch (2^15)
static constexpr int HWW = 65536;   // 256*256 grid cells
static constexpr int NGG = 3;
static constexpr int DD  = 4;

// ------------------------------------------------------- TDM (CDNA5) support
#if defined(__HIP_DEVICE_COMPILE__) && defined(__gfx1250__) && \
    __has_builtin(__builtin_amdgcn_tensor_load_to_lds)
#define HAVE_TDM 1
typedef __attribute__((ext_vector_type(4))) unsigned int v4u_t;
typedef __attribute__((ext_vector_type(4))) int          v4i_t;
typedef __attribute__((ext_vector_type(8))) int          v8i_t;

// 2-D bf16 tile load Global->LDS via Tensor Data Mover.
// LDS padding: 4 dwords inserted after every 16 dwords (pad codes 3/3),
// i.e. a 64B data row lands on an 80B (40 bf16) LDS pitch.
__device__ __forceinline__ void tdm_load_2d_bf16(unsigned lds_byte_off,
                                                 const void* gptr,
                                                 unsigned tile_k,     // dim0 elems
                                                 unsigned tile_rows,  // dim1 rows
                                                 unsigned row_stride) // elems
{
  unsigned long long ga = (unsigned long long)(uintptr_t)gptr;
  // D# group 0: count=1 | lds_addr | global_addr[56:0] | type=2
  v4u_t g0;
  g0.x = 1u;
  g0.y = lds_byte_off;
  g0.z = (unsigned)(ga & 0xFFFFFFFFu);
  g0.w = (unsigned)((ga >> 32) & 0x01FFFFFFu) | (2u << 30);
  // D# group 1
  unsigned td0 = row_stride;          // tensor_dim0
  unsigned td1 = 0x100000u;           // tensor_dim1 (generous, OOB-safe)
  v8i_t g1;
  g1[0] = (int)((1u << 16)            // data_size = 2 bytes
              | (1u << 20)            // pad_enable
              | (3u << 22)            // pad_interval: 16 dwords
              | (3u << 25));          // pad_amount : 4 dwords
  g1[1] = (int)((td0 & 0xFFFFu) << 16);
  g1[2] = (int)(((td0 >> 16) & 0xFFFFu) | ((td1 & 0xFFFFu) << 16));
  g1[3] = (int)(((td1 >> 16) & 0xFFFFu) | ((tile_k & 0xFFFFu) << 16));
  g1[4] = (int)(tile_rows & 0xFFFFu); // tile_dim1 (tile_dim2 = 0)
  g1[5] = (int)row_stride;            // tensor_dim0_stride[31:0]
  g1[6] = 0;
  g1[7] = 0;
  v4i_t gz4 = {0, 0, 0, 0};           // groups 2/3 unused (2-D tensor)
#if __clang_major__ >= 23
  v8i_t gz8 = {0, 0, 0, 0, 0, 0, 0, 0};
  __builtin_amdgcn_tensor_load_to_lds(g0, g1, gz4, gz4, gz8, 0);
#else
  __builtin_amdgcn_tensor_load_to_lds(g0, g1, gz4, gz4, 0);
#endif
}
#else
#define HAVE_TDM 0
#endif

// ---------------------------------------------------------------- zero fill
__global__ void k_zero(float* __restrict__ p, long long n) {
  long long i  = (long long)blockIdx.x * blockDim.x + threadIdx.x;
  long long st = (long long)gridDim.x * blockDim.x;
  for (; i < n; i += st) p[i] = 0.f;
}

// ------------------------------------------------------- cell counts (once)
__global__ void k_counts(const int* __restrict__ cell, const float* __restrict__ occ,
                         float* __restrict__ counts) {
  int i = blockIdx.x * 256 + threadIdx.x;          // i = (b*NG+g)*N + n
  int b = i / (NGG * NN);
  int n = i & (NN - 1);
  atomicAdd(&counts[(i / NN) * HWW + cell[i]], occ[b * NN + n]);
}

__global__ void k_weights(const int* __restrict__ cell, const float* __restrict__ occ,
                          const float* __restrict__ counts, float* __restrict__ wts) {
  int i = blockIdx.x * 256 + threadIdx.x;
  int b = i / (NGG * NN);
  int n = i & (NN - 1);
  float o   = occ[b * NN + n];
  float cnt = counts[(i / NN) * HWW + cell[i]];
  wts[i] = o / (cnt * o + 1e-6f);                  // occ/(cnt_pt+1e-6)
}

// -------------------------------------------- per-channel BN stats over B,N
__global__ __launch_bounds__(256) void k_bnstats(const float* __restrict__ t,
                                                 float* __restrict__ stat) {
  __shared__ float rs[256], rq[256];
  int c = blockIdx.x, tid = threadIdx.x;
  float s = 0.f, s2 = 0.f;
  for (int i = tid; i < BB * NN; i += 256) {
    int b = i >> 15;
    int n = i & (NN - 1);
    float x = t[((b * CC + c) << 15) + n];
    s += x; s2 += x * x;
  }
  rs[tid] = s; rq[tid] = s2;
  __syncthreads();
  for (int off = 128; off; off >>= 1) {
    if (tid < off) { rs[tid] += rs[tid + off]; rq[tid] += rq[tid + off]; }
    __syncthreads();
  }
  if (tid == 0) {
    float m = rs[0] * (1.f / (BB * NN));
    float v = rq[0] * (1.f / (BB * NN)) - m * m;
    stat[c] = m; stat[CC + c] = v;
  }
}

// ------------------------------------ fold BN into per-channel scale/shift
__global__ void k_smprep(const float* __restrict__ stat, const float* __restrict__ gamma,
                         const float* __restrict__ beta, float* __restrict__ st) {
  int c = threadIdx.x;
  float s = rsqrtf(stat[CC + c] + 1e-5f) * gamma[c];
  st[c]      = s;                     // scale
  st[CC + c] = beta[c] - stat[c] * s; // shift
}

// -------------------------------------------------- weighted scatter to grid
__global__ __launch_bounds__(256) void k_scatter(const float* __restrict__ tok,
                                                 const int* __restrict__ cell,
                                                 const float* __restrict__ wts,
                                                 const float* __restrict__ st,
                                                 float* __restrict__ grid, int g) {
  int n = blockIdx.x * 256 + threadIdx.x;
  int b = blockIdx.y;
  int ii = (b * NGG + g) * NN + n;
  int cl = cell[ii];
  float w = wts[ii];
  float* dst = grid + (size_t)(b * HWW + cl) * CC;
  const float* src = tok + ((size_t)b * CC) * NN + n;
  for (int c = 0; c < CC; ++c) {
    float x = src[(size_t)c * NN];
    atomicAdd(&dst[c], w * (x * st[c] + st[CC + c]));
  }
}

// ----------------- fused depthwise conv3x3 + bias + relu + conv3x3 + bias
// 16x16 output tile, 8-channel chunk, halo 2, both convs in LDS.
__global__ __launch_bounds__(256) void k_conv(const float* __restrict__ gin,
                                              float* __restrict__ gout,
                                              const float* __restrict__ k1,
                                              const float* __restrict__ b1,
                                              const float* __restrict__ k2,
                                              const float* __restrict__ b2) {
  __shared__ __align__(16) float sin_[20 * 20 * 8];
  __shared__ __align__(16) float smid[18 * 18 * 8];
  int tid = threadIdx.x;
  int ty = blockIdx.x >> 4, tx = blockIdx.x & 15;
  int h0 = ty * 16, w0 = tx * 16;
  int c0 = blockIdx.y * 8;
  int b  = blockIdx.z;
  const float* gbase = gin + ((size_t)b * HWW) * CC;
  for (int i = tid; i < 20 * 20 * 8; i += 256) {
    int ci = i & 7, p = i >> 3;
    int px = p % 20, py = p / 20;
    int hh = h0 + py - 2, ww = w0 + px - 2;
    float v = 0.f;
    if ((unsigned)hh < 256u && (unsigned)ww < 256u)
      v = gbase[(size_t)(hh * 256 + ww) * CC + c0 + ci];
    sin_[i] = v;
  }
  __syncthreads();
  for (int i = tid; i < 18 * 18 * 8; i += 256) {
    int ci = i & 7, p = i >> 3;
    int mx = p % 18, my = p / 18;
    float acc = b1[c0 + ci];
#pragma unroll
    for (int dy = 0; dy < 3; ++dy)
#pragma unroll
      for (int dx = 0; dx < 3; ++dx)
        acc += sin_[((my + dy) * 20 + mx + dx) * 8 + ci] * k1[(c0 + ci) * 9 + dy * 3 + dx];
    smid[i] = fmaxf(acc, 0.f);
  }
  __syncthreads();
  for (int i = tid; i < 16 * 16 * 8; i += 256) {
    int ci = i & 7, p = i >> 3;
    int ox = p & 15, oy = p >> 4;
    float acc = b2[c0 + ci];
#pragma unroll
    for (int dy = 0; dy < 3; ++dy)
#pragma unroll
      for (int dx = 0; dx < 3; ++dx)
        acc += smid[((oy + dy) * 18 + ox + dx) * 8 + ci] * k2[(c0 + ci) * 9 + dy * 3 + dx];
    gout[((size_t)b * HWW + (h0 + oy) * 256 + (w0 + ox)) * CC + c0 + ci] = acc;
  }
}

// ------------------------------------------------- gather + spatial residual
__global__ __launch_bounds__(256) void k_gather(float* __restrict__ tok,
                                                const float* __restrict__ grid,
                                                const int* __restrict__ cell,
                                                const float* __restrict__ occ,
                                                const float* __restrict__ scale, int g) {
  int n = blockIdx.x * 256 + threadIdx.x;
  int b = blockIdx.y;
  int cl = cell[(b * NGG + g) * NN + n];
  float o = occ[b * NN + n];
  const float* src = grid + (size_t)(b * HWW + cl) * CC;
  float* dst = tok + ((size_t)b * CC) * NN + n;
  for (int c = 0; c < CC; ++c)
    dst[(size_t)c * NN] += scale[c] * o * src[c];
}

// ---- fold BN into W1 (bf16), adjusted bias b1' = b1 + W1@(beta - m*s); W2->bf16
__global__ __launch_bounds__(256) void k_cmprep(const float* __restrict__ stat,
                                                const float* __restrict__ gamma,
                                                const float* __restrict__ beta,
                                                const float* __restrict__ w1,
                                                const float* __restrict__ b1,
                                                const float* __restrict__ w2,
                                                __bf16* __restrict__ w1bf,
                                                __bf16* __restrict__ w2bf,
                                                float* __restrict__ b1p) {
  __shared__ float red[256];
  int o = blockIdx.x, c = threadIdx.x;
  float s = rsqrtf(stat[CC + c] + 1e-5f) * gamma[c];
  float t = beta[c] - stat[c] * s;
  float w1v = w1[o * CC + c];
  w1bf[o * CC + c] = (__bf16)(w1v * s);
  w2bf[o * CC + c] = (__bf16)(w2[o * CC + c]);
  red[c] = w1v * t;
  __syncthreads();
  for (int off = 128; off; off >>= 1) {
    if (c < off) red[c] += red[c + off];
    __syncthreads();
  }
  if (c == 0) b1p[o] = b1[o] + red[0];
}

// --------------------------------------------------------- WMMA GEMM kernel
// D(256 x 32768) = W(256x256, bf16) * X(256x32768) per batch.
// 8 waves/block; block tile 128(M) x 64(N); K-step 32; wave tile 32x32 (4 WMMAs).
// A tile staged by the Tensor Data Mover (with D# LDS padding), B tile
// converted f32/bf16 -> bf16 and transposed to K-major by VALU.
template <bool SRC_BF16, bool RELU_OUT_BF16>
__global__ __launch_bounds__(256) void k_gemm(const __bf16* __restrict__ Wm,
                                              const void* __restrict__ Xsrc,
                                              void* __restrict__ Outv,
                                              const float* __restrict__ bias,
                                              const float* __restrict__ scale) {
  __shared__ __align__(16) __bf16 aT[128 * 40];  // [m][k], pitch 40 (16dw + 4dw pad)
  __shared__ __align__(16) __bf16 bT[64 * 40];   // [n][k], pitch 40
  int tid = threadIdx.x, lane = tid & 31, wid = tid >> 5;
  int lrow = lane & 15, lhi = lane >> 4;
  int wm = wid & 3, wn = wid >> 2;                // 4(M) x 2(N) wave grid
  int m0 = blockIdx.y * 128, n0 = blockIdx.x * 64, b = blockIdx.z;

  const __bf16* Xb = (const __bf16*)Xsrc + (size_t)b * CC * NN;
  const float*  Xf = (const float*)Xsrc + (size_t)b * CC * NN;

  v8f acc[2][2] = {};
  int sn = tid & 63, kq = tid >> 6;               // B staging coords

  for (int k0 = 0; k0 < CC; k0 += 32) {
    __syncthreads();
#if HAVE_TDM
    // stage A via TDM: 128 rows x 32 bf16, 80B LDS pitch via D# padding.
    if (wid == 0)
      tdm_load_2d_bf16((unsigned)(uintptr_t)&aT[0],
                       Wm + (size_t)m0 * CC + k0, 32u, 128u, (unsigned)CC);
#else
    for (int ch = tid; ch < 512; ch += 256) {
      int row = ch >> 2, kk = (ch & 3) * 8;
      *(uint4*)&aT[row * 40 + kk] = *(const uint4*)(Wm + (m0 + row) * CC + k0 + kk);
    }
#endif
    // stage B (activations, transpose to [n][k], convert to bf16)
#pragma unroll
    for (int kr = 0; kr < 8; ++kr) {
      int k = kr * 4 + kq;
      float x;
      if (SRC_BF16) x = (float)Xb[(size_t)(k0 + k) * NN + n0 + sn];
      else          x = Xf[(size_t)(k0 + k) * NN + n0 + sn];
      bT[sn * 40 + k] = (__bf16)x;
    }
    // prefetch next K-tile of B (global_prefetch_b8)
    if (k0 + 32 < CC) {
      if (SRC_BF16) __builtin_prefetch(&Xb[(size_t)(k0 + 32 + kq) * NN + n0 + sn], 0, 1);
      else          __builtin_prefetch(&Xf[(size_t)(k0 + 32 + kq) * NN + n0 + sn], 0, 1);
    }
#if HAVE_TDM
    // Drain TDM before publishing the tile (no-op for non-issuing waves).
    __builtin_amdgcn_s_wait_tensorcnt(0);
#endif
    __syncthreads();

    union { v16bf v; unsigned u[8]; } afr[2], bfr[2];
    // A fragment: lo lanes K{0-7,16-23}, hi lanes K{8-15,24-31} (ISA 16-bit A layout)
#pragma unroll
    for (int mi = 0; mi < 2; ++mi) {
      int row = wm * 32 + mi * 16 + lrow;
#pragma unroll
      for (int p = 0; p < 8; ++p) {
        int kk = (p < 4 ? 2 * p : 8 + 2 * p) + lhi * 8;
        afr[mi].u[p] = *(const unsigned*)&aT[row * 40 + kk];
      }
    }
    // B fragment: lo lanes K{0-7,16-23}, hi lanes K{8-15,24-31} (matches A layout)
#pragma unroll
    for (int ni = 0; ni < 2; ++ni) {
      int col = wn * 32 + ni * 16 + lrow;
#pragma unroll
      for (int p = 0; p < 8; ++p) {
        int kk = (p < 4 ? 2 * p : 8 + 2 * p) + lhi * 8;
        bfr[ni].u[p] = *(const unsigned*)&bT[col * 40 + kk];
      }
    }
#pragma unroll
    for (int mi = 0; mi < 2; ++mi)
#pragma unroll
      for (int ni = 0; ni < 2; ++ni)
        acc[mi][ni] = __builtin_amdgcn_wmma_f32_16x16x32_bf16(
            false, afr[mi].v, false, bfr[ni].v, (short)0, acc[mi][ni], false, false);
  }

  // epilogue: C/D layout -> M = r + 8*laneHi, N = lane%16
#pragma unroll
  for (int mi = 0; mi < 2; ++mi)
#pragma unroll
    for (int ni = 0; ni < 2; ++ni) {
      int gn = n0 + wn * 32 + ni * 16 + lrow;
#pragma unroll
      for (int r = 0; r < 8; ++r) {
        int gm = m0 + wm * 32 + mi * 16 + r + lhi * 8;
        float v = acc[mi][ni][r] + bias[gm];
        if (RELU_OUT_BF16) {
          ((__bf16*)Outv)[(size_t)b * CC * NN + (size_t)gm * NN + gn] = (__bf16)fmaxf(v, 0.f);
        } else {
          float* op = (float*)Outv + (size_t)b * CC * NN + (size_t)gm * NN + gn;
          *op += scale[gm] * v;
        }
      }
    }
}

// ---------------------------------------------------------------------------
extern "C" void kernel_launch(void* const* d_in, const int* in_sizes, int n_in,
                              void* d_out, int out_size, void* d_ws, size_t ws_size,
                              hipStream_t stream) {
  const float* tokens_in = (const float*)d_in[0];
  const int*   cell_ind  = (const int*)d_in[1];
  const float* occ       = (const float*)d_in[2];
  const float* cm_gamma  = (const float*)d_in[3];
  const float* cm_beta   = (const float*)d_in[4];
  const float* cm_w1     = (const float*)d_in[5];
  const float* cm_b1     = (const float*)d_in[6];
  const float* cm_w2     = (const float*)d_in[7];
  const float* cm_b2     = (const float*)d_in[8];
  const float* cm_scale  = (const float*)d_in[9];
  const float* sm_gamma  = (const float*)d_in[10];
  const float* sm_beta   = (const float*)d_in[11];
  const float* sm_k1     = (const float*)d_in[12];
  const float* sm_b1     = (const float*)d_in[13];
  const float* sm_k2     = (const float*)d_in[14];
  const float* sm_b2     = (const float*)d_in[15];
  const float* sm_scale  = (const float*)d_in[16];
  (void)in_sizes; (void)n_in; (void)out_size; (void)ws_size;

  float* tok = (float*)d_out;              // working tokens, updated in place
  char* ws = (char*)d_ws;
  size_t off = 0;
  float* grid1  = (float*)(ws + off); off += (size_t)BB * HWW * CC * 4;  // 128 MB
  float* grid2  = (float*)(ws + off); off += (size_t)BB * HWW * CC * 4;  // 128 MB
  float* counts = (float*)(ws + off); off += (size_t)BB * NGG * HWW * 4;
  float* wts    = (float*)(ws + off); off += (size_t)BB * NGG * NN * 4;
  float* stat   = (float*)(ws + off); off += 4 * CC * 4;   // m,v + s,t
  float* b1p    = (float*)(ws + off); off += CC * 4;
  __bf16* W1bf  = (__bf16*)(ws + off); off += (size_t)CC * CC * 2;
  __bf16* W2bf  = (__bf16*)(ws + off); off += (size_t)CC * CC * 2;
  __bf16* hbf   = (__bf16*)grid1;  // alias: grid1 is free by the time GEMM1 runs

  hipMemcpyAsync(tok, tokens_in, (size_t)BB * CC * NN * 4,
                 hipMemcpyDeviceToDevice, stream);

  // cell counts / weights (fixed across layers)
  k_zero<<<1024, 256, 0, stream>>>(counts, (long long)BB * NGG * HWW);
  k_counts<<<BB * NGG * NN / 256, 256, 0, stream>>>(cell_ind, occ, counts);
  k_weights<<<BB * NGG * NN / 256, 256, 0, stream>>>(cell_ind, occ, counts, wts);

  for (int d = 0; d < DD; ++d) {
    int g = d % NGG;
    // ---- spatial mix ----
    k_bnstats<<<CC, 256, 0, stream>>>(tok, stat);
    k_smprep<<<1, 256, 0, stream>>>(stat, sm_gamma + d * CC, sm_beta + d * CC, stat + 2 * CC);
    k_zero<<<8192, 256, 0, stream>>>(grid1, (long long)BB * HWW * CC);
    k_scatter<<<dim3(NN / 256, BB), 256, 0, stream>>>(tok, cell_ind, wts, stat + 2 * CC, grid1, g);
    k_conv<<<dim3(256, CC / 8, BB), 256, 0, stream>>>(grid1, grid2,
        sm_k1 + (size_t)d * CC * 9, sm_b1 + d * CC,
        sm_k2 + (size_t)d * CC * 9, sm_b2 + d * CC);
    k_gather<<<dim3(NN / 256, BB), 256, 0, stream>>>(tok, grid2, cell_ind, occ,
                                                     sm_scale + d * CC, g);
    // ---- channel mix (WMMA) ----
    k_bnstats<<<CC, 256, 0, stream>>>(tok, stat);
    k_cmprep<<<CC, 256, 0, stream>>>(stat, cm_gamma + d * CC, cm_beta + d * CC,
                                     cm_w1 + (size_t)d * CC * CC, cm_b1 + d * CC,
                                     cm_w2 + (size_t)d * CC * CC, W1bf, W2bf, b1p);
    k_gemm<false, true><<<dim3(NN / 64, CC / 128, BB), 256, 0, stream>>>(
        W1bf, tok, hbf, b1p, nullptr);
    k_gemm<true, false><<<dim3(NN / 64, CC / 128, BB), 256, 0, stream>>>(
        W2bf, hbf, tok, cm_b2 + d * CC, cm_scale + d * CC);
  }
}
